// EfficientAdditiveAttention_4054449128160
// MI455X (gfx1250) — matrix operand order repaired and checked by reading, not compiled, not run on hardware
//
#include <hip/hip_runtime.h>

typedef __attribute__((ext_vector_type(16))) __bf16 v16bf;
typedef __attribute__((ext_vector_type(8)))  float  v8f;
typedef unsigned short u16;
typedef unsigned int   u32;

#define B_BATCH 64
#define NTOK    784
#define NTILES  49
#define CDIM    512
#define SCALE_F 0.04419417382415922f   // 512^-0.5
#define EPS_F   1e-12f

struct alignas(16) U128 { u32 a, b, c, d; };
union BFrag { v16bf v; u16 u[16]; U128 q[2]; };

__device__ __forceinline__ u16 f2bf(float f) {
  u32 x = __float_as_uint(f);
  x += 0x7FFFu + ((x >> 16) & 1u);   // round-to-nearest-even
  return (u16)(x >> 16);
}
__device__ __forceinline__ float bf2f(u16 h) {
  return __uint_as_float(((u32)h) << 16);
}
__device__ __forceinline__ float red16(float v) {
#pragma unroll
  for (int m = 1; m < 16; m <<= 1) v += __shfl_xor(v, m, 32);
  return v;
}

// ---- kernel 0: transpose + convert four 512x512 f32 weights to bf16 W^T ----
__global__ __launch_bounds__(256) void eaa_convert_w(
    const float* __restrict__ Wq, const float* __restrict__ Wk,
    const float* __restrict__ Wp, const float* __restrict__ Wf,
    u16* __restrict__ wt) {
  int idx = blockIdx.x * 256 + threadIdx.x;    // 4 * 512 * 512 total
  int mat = idx >> 18;
  int k   = (idx >> 9) & 511;
  int d   = idx & 511;
  const float* W = (mat == 0) ? Wq : (mat == 1) ? Wk : (mat == 2) ? Wp : Wf;
  wt[((size_t)mat << 18) + ((size_t)d << 9) + k] = f2bf(W[((size_t)k << 9) + d]);
}

// ---- kernel 1: query/key GEMMs + row l2norm + g = q . w_g ----
__global__ __launch_bounds__(256) void eaa_qk(
    const float* __restrict__ x,
    const u16* __restrict__ WqT, const float* __restrict__ bq,
    const u16* __restrict__ WkT, const float* __restrict__ bk,
    const float* __restrict__ wg,
    u16* __restrict__ qout, u16* __restrict__ kout, float* __restrict__ gout) {
  __shared__ u16  tile[16 * 512];            // 16KB: x-tile, then q, then k (bf16)
  __shared__ float partq[16][8], partk[16][8], partg[16][8];

  const int tid  = threadIdx.x;
  const int lane = tid & 31;
  const int wave = tid >> 5;
  const int l16  = lane & 15;
  const int h    = lane >> 4;
  const int b    = blockIdx.x / NTILES;
  const int n0   = (blockIdx.x % NTILES) * 16;

  // stage t tile [n][c]: t[n,c] = x[b, c, n0+n]
  const float* xb = x + (size_t)b * CDIM * NTOK + n0;
  for (int idx = tid; idx < 16 * 512; idx += 256) {
    int c = idx >> 4, n = idx & 15;
    tile[n * 512 + c] = f2bf(xb[(size_t)c * NTOK + n]);
  }
  __syncthreads();

  v8f accq[4] = {};
  v8f acck[4] = {};
  for (int ks = 0; ks < 16; ++ks) {
    BFrag a;   // A 16x32 bf16: lane row=l16; K = ks*32 + {8h..8h+7, 8h+16..8h+23}
    const u16* ap = &tile[l16 * 512 + ks * 32 + h * 8];
    a.q[0] = *(const U128*)(ap);
    a.q[1] = *(const U128*)(ap + 16);
#pragma unroll
    for (int ct = 0; ct < 4; ++ct) {
      const int d = wave * 64 + ct * 16 + l16;
      const u16* bpq = WqT + ((size_t)d << 9) + ks * 32 + h * 8;  // K = ks*32 + {8h..8h+7, 8h+16..8h+23}
      BFrag fb;
      fb.q[0] = *(const U128*)(bpq);
      fb.q[1] = *(const U128*)(bpq + 16);
      accq[ct] = __builtin_amdgcn_wmma_f32_16x16x32_bf16(
          false, a.v, false, fb.v, (short)0, accq[ct], false, false);
      const u16* bpk = WkT + ((size_t)d << 9) + ks * 32 + h * 8;
      fb.q[0] = *(const U128*)(bpk);
      fb.q[1] = *(const U128*)(bpk + 16);
      acck[ct] = __builtin_amdgcn_wmma_f32_16x16x32_bf16(
          false, a.v, false, fb.v, (short)0, acck[ct], false, false);
    }
  }

  // bias add + per-row sum of squares (row = r + 8h)
  float psq[8] = {0,0,0,0,0,0,0,0};
  float psk[8] = {0,0,0,0,0,0,0,0};
#pragma unroll
  for (int ct = 0; ct < 4; ++ct) {
    const int d = wave * 64 + ct * 16 + l16;
    const float biasq = bq[d], biask = bk[d];
#pragma unroll
    for (int r = 0; r < 8; ++r) {
      float vq = accq[ct][r] + biasq; accq[ct][r] = vq; psq[r] += vq * vq;
      float vk = acck[ct][r] + biask; acck[ct][r] = vk; psk[r] += vk * vk;
    }
  }
#pragma unroll
  for (int r = 0; r < 8; ++r) {
    float sq = red16(psq[r]);
    float sk = red16(psk[r]);
    if (l16 == 0) { partq[r + 8 * h][wave] = sq; partk[r + 8 * h][wave] = sk; }
  }
  __syncthreads();   // K-loop reads of tile done; cross-wave partials visible

  float iq[8], ik[8];
#pragma unroll
  for (int r = 0; r < 8; ++r) {
    const int row = r + 8 * h;
    float sq = 0.f, sk = 0.f;
#pragma unroll
    for (int wv = 0; wv < 8; ++wv) { sq += partq[row][wv]; sk += partk[row][wv]; }
    iq[r] = 1.0f / fmaxf(sqrtf(sq), EPS_F);
    ik[r] = 1.0f / fmaxf(sqrtf(sk), EPS_F);
  }

  // normalize, compute g partials, write q (bf16) to tile
  float pg[8] = {0,0,0,0,0,0,0,0};
#pragma unroll
  for (int ct = 0; ct < 4; ++ct) {
    const int d = wave * 64 + ct * 16 + l16;
    const float w = wg[d];
#pragma unroll
    for (int r = 0; r < 8; ++r) {
      float qv = accq[ct][r] * iq[r];
      acck[ct][r] *= ik[r];
      pg[r] += qv * w;
      tile[(r + 8 * h) * 512 + d] = f2bf(qv);
    }
  }
#pragma unroll
  for (int r = 0; r < 8; ++r) {
    float sg = red16(pg[r]);
    if (l16 == 0) partg[r + 8 * h][wave] = sg;
  }
  __syncthreads();

  if (tid < 16) {
    float s = 0.f;
#pragma unroll
    for (int wv = 0; wv < 8; ++wv) s += partg[tid][wv];
    gout[(size_t)b * NTOK + n0 + tid] = s;
  }
  {
    U128* qo = (U128*)(qout + ((size_t)b * NTOK + n0) * CDIM);
    const U128* ts = (const U128*)tile;
    for (int i = tid; i < 1024; i += 256) qo[i] = ts[i];
  }
  __syncthreads();

#pragma unroll
  for (int ct = 0; ct < 4; ++ct) {
    const int d = wave * 64 + ct * 16 + l16;
#pragma unroll
    for (int r = 0; r < 8; ++r) tile[(r + 8 * h) * 512 + d] = f2bf(acck[ct][r]);
  }
  __syncthreads();
  {
    U128* ko = (U128*)(kout + ((size_t)b * NTOK + n0) * CDIM);
    const U128* ts = (const U128*)tile;
    for (int i = tid; i < 1024; i += 256) ko[i] = ts[i];
  }
}

// ---- kernel 2: attn = l2norm(g*SCALE over tokens); ctx = attn^T @ q ----
__global__ __launch_bounds__(256) void eaa_ctx(
    const float* __restrict__ g, const u16* __restrict__ q, float* __restrict__ ctx) {
  __shared__ float red[256];
  __shared__ float attn[NTOK];
  const int b = blockIdx.x, tid = threadIdx.x;
  const float* gb = g + (size_t)b * NTOK;
  float s = 0.f;
  for (int n = tid; n < NTOK; n += 256) { float v = gb[n] * SCALE_F; s += v * v; }
  red[tid] = s;
  __syncthreads();
  for (int off = 128; off > 0; off >>= 1) {
    if (tid < off) red[tid] += red[tid + off];
    __syncthreads();
  }
  const float inv = 1.0f / fmaxf(sqrtf(red[0]), EPS_F);
  for (int n = tid; n < NTOK; n += 256) attn[n] = gb[n] * SCALE_F * inv;
  __syncthreads();
  const u16* qb = q + (size_t)b * NTOK * CDIM;
#pragma unroll
  for (int dd = 0; dd < 2; ++dd) {
    const int d = tid * 2 + dd;
    float acc = 0.f;
    for (int n = 0; n < NTOK; ++n) acc += attn[n] * bf2f(qb[(size_t)n * CDIM + d]);
    ctx[(size_t)b * CDIM + d] = acc;
  }
}

// ---- kernel 3: out = ((ctx*k)@Wp + bp + q) @ Wf + bf, transposed store ----
__global__ __launch_bounds__(256) void eaa_out(
    const u16* __restrict__ kin, const u16* __restrict__ qin, const float* __restrict__ ctx,
    const u16* __restrict__ WpT, const float* __restrict__ bp,
    const u16* __restrict__ WfT, const float* __restrict__ bfv,
    float* __restrict__ out) {
  __shared__ float smemf[16 * 512];          // 32KB, aliased as bf16 tile
  u16* tile = (u16*)smemf;

  const int tid  = threadIdx.x;
  const int lane = tid & 31;
  const int wave = tid >> 5;
  const int l16  = lane & 15;
  const int h    = lane >> 4;
  const int b    = blockIdx.x / NTILES;
  const int n0   = (blockIdx.x % NTILES) * 16;
  const size_t base = ((size_t)b * NTOK + n0) * CDIM;
  const float* ctxb = ctx + (size_t)b * CDIM;

  // stage A1 = ctx (*) k  as bf16 [n][c]
  for (int idx = tid; idx < 8192; idx += 256) {
    tile[idx] = f2bf(bf2f(kin[base + idx]) * ctxb[idx & 511]);
  }
  __syncthreads();

  v8f acc[4] = {};
  for (int ks = 0; ks < 16; ++ks) {
    BFrag a;
    const u16* ap = &tile[l16 * 512 + ks * 32 + h * 8];
    a.q[0] = *(const U128*)(ap);
    a.q[1] = *(const U128*)(ap + 16);
#pragma unroll
    for (int ct = 0; ct < 4; ++ct) {
      const int d = wave * 64 + ct * 16 + l16;
      const u16* bpp = WpT + ((size_t)d << 9) + ks * 32 + h * 8;
      BFrag fb;
      fb.q[0] = *(const U128*)(bpp);
      fb.q[1] = *(const U128*)(bpp + 16);
      acc[ct] = __builtin_amdgcn_wmma_f32_16x16x32_bf16(
          false, a.v, false, fb.v, (short)0, acc[ct], false, false);
    }
  }
  // + bp + query residual
#pragma unroll
  for (int ct = 0; ct < 4; ++ct) {
    const int d = wave * 64 + ct * 16 + l16;
    const float bpv = bp[d];
#pragma unroll
    for (int r = 0; r < 8; ++r) {
      const int row = r + 8 * h;
      acc[ct][r] += bpv + bf2f(qin[base + (size_t)row * CDIM + d]);
    }
  }
  __syncthreads();
#pragma unroll
  for (int ct = 0; ct < 4; ++ct) {
    const int d = wave * 64 + ct * 16 + l16;
#pragma unroll
    for (int r = 0; r < 8; ++r) tile[(r + 8 * h) * 512 + d] = f2bf(acc[ct][r]);
  }
  __syncthreads();

  v8f acc2[4] = {};
  for (int ks = 0; ks < 16; ++ks) {
    BFrag a;
    const u16* ap = &tile[l16 * 512 + ks * 32 + h * 8];
    a.q[0] = *(const U128*)(ap);
    a.q[1] = *(const U128*)(ap + 16);
#pragma unroll
    for (int ct = 0; ct < 4; ++ct) {
      const int d = wave * 64 + ct * 16 + l16;
      const u16* bpp = WfT + ((size_t)d << 9) + ks * 32 + h * 8;
      BFrag fb;
      fb.q[0] = *(const U128*)(bpp);
      fb.q[1] = *(const U128*)(bpp + 16);
      acc2[ct] = __builtin_amdgcn_wmma_f32_16x16x32_bf16(
          false, a.v, false, fb.v, (short)0, acc2[ct], false, false);
    }
  }
  __syncthreads();   // all bf16 tile reads done before f32 overwrite
#pragma unroll
  for (int ct = 0; ct < 4; ++ct) {
    const int d = wave * 64 + ct * 16 + l16;
    const float bv = bfv[d];
#pragma unroll
    for (int r = 0; r < 8; ++r) smemf[d * 16 + (r + 8 * h)] = acc2[ct][r] + bv;
  }
  __syncthreads();

  // coalesced transposed store: out[b, d, n0+n]
  float* ob = out + (size_t)b * CDIM * NTOK + n0;
  for (int idx = tid; idx < 8192; idx += 256) {
    ob[(size_t)(idx >> 4) * NTOK + (idx & 15)] = smemf[idx];
  }
}

extern "C" void kernel_launch(void* const* d_in, const int* in_sizes, int n_in,
                              void* d_out, int out_size, void* d_ws, size_t ws_size,
                              hipStream_t stream) {
  const float* x  = (const float*)d_in[0];
  const float* Wq = (const float*)d_in[1];
  const float* bq = (const float*)d_in[2];
  const float* Wk = (const float*)d_in[3];
  const float* bk = (const float*)d_in[4];
  const float* wg = (const float*)d_in[5];
  const float* Wp = (const float*)d_in[6];
  const float* bp = (const float*)d_in[7];
  const float* Wf = (const float*)d_in[8];
  const float* bf = (const float*)d_in[9];
  float* out = (float*)d_out;

  u16* wt   = (u16*)d_ws;                          // 4 x 512x512 bf16 W^T
  u16* WqT  = wt;
  u16* WkT  = wt + ((size_t)1 << 18);
  u16* WpT  = wt + ((size_t)2 << 18);
  u16* WfT  = wt + ((size_t)3 << 18);
  u16* qbuf = wt + ((size_t)4 << 18);              // [B,N,D] bf16
  u16* kbuf = qbuf + (size_t)B_BATCH * NTOK * CDIM;
  float* gbuf = (float*)(kbuf + (size_t)B_BATCH * NTOK * CDIM);  // [B,N]
  float* cbuf = gbuf + (size_t)B_BATCH * NTOK;                   // [B,D]

  eaa_convert_w<<<4096, 256, 0, stream>>>(Wq, Wk, Wp, Wf, wt);
  eaa_qk<<<B_BATCH * NTILES, 256, 0, stream>>>(x, WqT, bq, WkT, bk, wg,
                                               qbuf, kbuf, gbuf);
  eaa_ctx<<<B_BATCH, 256, 0, stream>>>(gbuf, qbuf, cbuf);
  eaa_out<<<B_BATCH * NTILES, 256, 0, stream>>>(kbuf, qbuf, cbuf, WpT, bp,
                                                WfT, bf, out);
}
